// PopSpikeActor_69415261438437
// MI455X (gfx1250) — matrix operand order repaired and checked by reading, not compiled, not run on hardware
//
#include <hip/hip_runtime.h>
#include <hip/hip_bf16.h>

typedef _Float16 v16h __attribute__((ext_vector_type(16)));
typedef _Float16 v8h  __attribute__((ext_vector_type(8)));
typedef _Float16 v4h  __attribute__((ext_vector_type(4)));
typedef float    v8f  __attribute__((ext_vector_type(8)));

#define OBS_DIM 256
#define POP_DIM 10
#define N_IN    2560          // OBS_DIM * POP_DIM
#define HID     256
#define OUTPOP  80            // ACT_DIM * DE_POP
#define ACT_DIM 8
#define TS      25
#define BT      16            // batch rows per workgroup (one WMMA M tile)
#define ENC_VTH 0.999f

#define W2P_BYTES ((HID / 32) * (HID / 16) * 512 * 2)     // 131072
#define W3P_BYTES ((HID / 32) * (OUTPOP / 16) * 512 * 2)  // 40960

// ---------------------------------------------------------------------------
// Phase A: population encoding + 25-step encoder spike loop, bit-packed.
// spk[b*N_IN + k] holds s_t in bit t (t = 0..24).
// ---------------------------------------------------------------------------
__global__ void enc_pack_kernel(const float* __restrict__ obs,
                                const float* __restrict__ mean,
                                const float* __restrict__ stdv,
                                unsigned int* __restrict__ spk, int total)
{
    int idx = blockIdx.x * blockDim.x + threadIdx.x;
    if (idx >= total) return;
    int b = idx / N_IN;
    int k = idx - b * N_IN;        // k = o*POP_DIM + p
    int o = k / POP_DIM;
    float m = mean[k];
    float s = stdv[k];
    float d = obs[b * OBS_DIM + o] - m;
    float a = __expf(-0.5f * d * d / (s * s));
    float volt = 0.f;
    unsigned int bits = 0u;
#pragma unroll
    for (int t = 0; t < TS; ++t) {
        volt += a;
        if (volt > ENC_VTH) { bits |= (1u << t); volt -= ENC_VTH; }
    }
    spk[idx] = bits;
}

// ---------------------------------------------------------------------------
// Phase A2: pack fp32 weights W[H][K] (row-major) into f16 WMMA B-fragments.
//   n = ht*16 + (lane & 15)
//   k = kc*32 + (i < 8 ? 8h + i : 16 + 8h + (i - 8)), h = lane >= 16
//   (B: lanes 0-15 hold K 0..7,16..23; lanes 16-31 hold K 8..15,24..31)
//   B[k][n] = W[n*K + k]
// ---------------------------------------------------------------------------
__global__ void pack_w_kernel(const float* __restrict__ W, _Float16* __restrict__ dst,
                              int K, int H, int total)
{
    int idx = blockIdx.x * blockDim.x + threadIdx.x;
    if (idx >= total) return;
    int i    = idx & 15;
    int lane = (idx >> 4) & 31;
    int fi   = idx >> 9;
    int NT   = H >> 4;
    int kc   = fi / NT;
    int ht   = fi - kc * NT;
    int n = ht * 16 + (lane & 15);
    int k = kc * 32 + ((i & 8) << 1) + ((lane & 16) >> 1) + (i & 7);
    dst[idx] = (_Float16)W[n * K + k];
}

// ---------------------------------------------------------------------------
// Async DMA: one 16-byte global->LDS transfer per lane (ASYNCcnt-tracked).
// VDST VGPR carries the LDS byte address (low 32 bits of the generic pointer:
// flat->LDS mapping truncates to addr[31:0], ISA 10.2).
// ---------------------------------------------------------------------------
__device__ __forceinline__ void async_copy_b128(unsigned lds_addr,
                                                unsigned long long gbase,
                                                unsigned goff)
{
    asm volatile("global_load_async_to_lds_b128 %0, %1, %2"
                 :: "v"(lds_addr), "v"(goff), "s"(gbase)
                 : "memory");
}

__device__ __forceinline__ void wait_asynccnt0()
{
    asm volatile("s_wait_asynccnt 0" ::: "memory");
}

// ---------------------------------------------------------------------------
// Wave-level GEMM tile: 16x16 f32 accumulator over K in chunks of 32.
// A: 16 rows of f16 spikes in LDS (row-major, stride ldA halves).
//    lane<16  -> row=lane,    k = {0..7} u {16..23}
//    lane>=16 -> row=lane-16, k = {8..15} u {24..31}
// B: pre-packed fragments (global for W1, LDS for W2/W3), 32B/lane/chunk.
// ---------------------------------------------------------------------------
template <bool PF>
__device__ __forceinline__ v8f wave_gemm(const _Float16* __restrict__ Alds, int ldA,
                                         const _Float16* __restrict__ Bfrag, int bstride,
                                         int nChunks, int lane, v8f acc)
{
    const _Float16* ap = Alds + (lane & 15) * ldA + ((lane & 16) ? 8 : 0);
    const _Float16* bp = Bfrag + lane * 16;
#pragma unroll 4
    for (int kc = 0; kc < nChunks; ++kc) {
        union { v16h v; v8h h[2]; } au;
        au.h[0] = *(const v8h*)(ap);
        au.h[1] = *(const v8h*)(ap + 16);
        v16h bv = *(const v16h*)(bp);
        if (PF) __builtin_prefetch(bp + bstride, 0, 0);   // next W1 chunk -> GL2
        acc = __builtin_amdgcn_wmma_f32_16x16x32_f16(false, au.v, false, bv,
                                                     (short)0, acc, false, false);
        ap += 32;
        bp += bstride;
    }
    return acc;
}

// LIF with dynamic threshold, elementwise on the 16x16 tile in C/D VGPR layout.
__device__ __forceinline__ void lif_update(const v8f& acc, float bias,
                                           v8f& c, v8f& v, v8f& s)
{
#pragma unroll
    for (int j = 0; j < 8; ++j) {
        float cj = c[j] * 0.5f + acc[j] + bias;
        float vprev = v[j];
        float vj = vprev * 0.75f * (1.f - s[j]) + cj;
        float vth = 0.25f + 0.5f * (__expf((vprev - vj) * (1.f / 3.f)) - 1.f);
        c[j] = cj;
        v[j] = vj;
        s[j] = (vj > vth) ? 1.f : 0.f;
    }
}

// Scatter spikes (C/D layout: row = j + 8*(lane>=16), col = ht*16 + (lane&15))
__device__ __forceinline__ void store_spikes(_Float16* buf, int ldB, int ht, int lane,
                                             const v8f& s)
{
    int col = ht * 16 + (lane & 15);
    int rbase = (lane & 16) ? 8 : 0;
#pragma unroll
    for (int j = 0; j < 8; ++j)
        buf[(rbase + j) * ldB + col] = (_Float16)s[j];
}

// ---------------------------------------------------------------------------
// Phase B: the whole 25-step recurrent spike MLP. One workgroup = 16 batch
// rows; 16 waves = 16 hidden-column tiles. LIF state lives in VGPRs; W2/W3
// fragments are DMA'd into LDS once and reused for all 25 timesteps.
// ---------------------------------------------------------------------------
__global__ void __launch_bounds__(512)
snn_mlp_kernel(const unsigned int* __restrict__ spk,
               const _Float16* __restrict__ w1p,
               const _Float16* __restrict__ w2p,
               const _Float16* __restrict__ w3p,
               const float* __restrict__ b1,
               const float* __restrict__ b2,
               const float* __restrict__ b3,
               const float* __restrict__ dec_w,
               const float* __restrict__ dec_b,
               float* __restrict__ out)
{
    extern __shared__ char smem[];
    _Float16* sp  = (_Float16*)smem;                                  // [BT][N_IN]  80 KB
    _Float16* s1b = (_Float16*)(smem + (size_t)BT * N_IN * 2);        // [BT][HID]    8 KB
    _Float16* s2b = s1b + (size_t)BT * HID;                           // [BT][HID]    8 KB
    float*    pop = (float*)(s2b + (size_t)BT * HID);                 // [BT][OUTPOP] 5 KB
    _Float16* w2l = (_Float16*)((char*)pop + (size_t)BT * OUTPOP * 4); // 128 KB
    _Float16* w3l = (_Float16*)((char*)w2l + W2P_BYTES);               //  40 KB

    int tid   = threadIdx.x;
    int lane  = tid & 31;
    int wave  = tid >> 5;                  // 0..15
    int bbase = blockIdx.x * BT;

    // ---- stage packed W2/W3 into LDS via async DMA (once, reused 25x) ----
    {
        unsigned w2_lds = (unsigned)(uintptr_t)w2l;
        unsigned w3_lds = (unsigned)(uintptr_t)w3l;
        unsigned long long g2 = (unsigned long long)(uintptr_t)w2p;
        unsigned long long g3 = (unsigned long long)(uintptr_t)w3p;
#pragma unroll
        for (int i = 0; i < W2P_BYTES / (512 * 16); ++i) {          // 16 iters
            unsigned off = (unsigned)(i * 512 + tid) * 16u;
            async_copy_b128(w2_lds + off, g2, off);
        }
#pragma unroll
        for (int i = 0; i < W3P_BYTES / (512 * 16); ++i) {          // 5 iters
            unsigned off = (unsigned)(i * 512 + tid) * 16u;
            async_copy_b128(w3_lds + off, g3, off);
        }
        wait_asynccnt0();
    }

    float bias1 = b1[wave * 16 + (lane & 15)];
    float bias2 = b2[wave * 16 + (lane & 15)];
    float bias3 = (wave < 5) ? b3[wave * 16 + (lane & 15)] : 0.f;

    v8f c1 = {}, v1 = {}, s1 = {};
    v8f c2 = {}, v2 = {}, s2 = {};
    v8f c3 = {}, v3 = {}, s3 = {}, accsum = {};
    v8f zero = {};

    const uint4* spw4 = (const uint4*)(spk + (size_t)bbase * N_IN);
    __syncthreads();   // LDS weight staging visible to all waves

    for (int t = 0; t < TS; ++t) {
        // unpack this timestep's encoder spikes into LDS as f16 0/1 (4-wide)
        for (int e = tid; e < BT * N_IN / 4; e += 512) {
            uint4 w = spw4[e];
            v4h h;
            h[0] = (_Float16)((w.x >> t) & 1u);
            h[1] = (_Float16)((w.y >> t) & 1u);
            h[2] = (_Float16)((w.z >> t) & 1u);
            h[3] = (_Float16)((w.w >> t) & 1u);
            *(v4h*)(sp + 4 * e) = h;
        }
        __syncthreads();

        // layer 1: [BT x 2560] x [2560 x 256], B streamed from L2 w/ prefetch
        v8f acc = wave_gemm<true>(sp, N_IN, w1p + wave * 512, 16 * 512,
                                  N_IN / 32, lane, zero);
        lif_update(acc, bias1, c1, v1, s1);
        store_spikes(s1b, HID, wave, lane, s1);
        __syncthreads();

        // layer 2: [BT x 256] x [256 x 256], B from LDS
        acc = wave_gemm<false>(s1b, HID, w2l + wave * 512, 16 * 512,
                               HID / 32, lane, zero);
        lif_update(acc, bias2, c2, v2, s2);
        store_spikes(s2b, HID, wave, lane, s2);
        __syncthreads();

        // layer 3: [BT x 256] x [256 x 80], B from LDS (waves 0..4)
        if (wave < 5) {
            acc = wave_gemm<false>(s2b, HID, w3l + wave * 512, 5 * 512,
                                   HID / 32, lane, zero);
            lif_update(acc, bias3, c3, v3, s3);
#pragma unroll
            for (int j = 0; j < 8; ++j) accsum[j] += s3[j];
        }
        __syncthreads();
    }

    // rate decode to LDS
    if (wave < 5) {
        int col = wave * 16 + (lane & 15);
        int rbase = (lane & 16) ? 8 : 0;
#pragma unroll
        for (int j = 0; j < 8; ++j)
            pop[(rbase + j) * OUTPOP + col] = accsum[j] * (1.f / (float)TS);
    }
    __syncthreads();

    // grouped decoder + tanh: 16 rows x 8 actions
    if (tid < BT * ACT_DIM) {
        int r = tid >> 3, a = tid & 7;
        float raw = dec_b[a];
#pragma unroll
        for (int p = 0; p < POP_DIM; ++p)
            raw += pop[r * OUTPOP + a * POP_DIM + p] * dec_w[a * POP_DIM + p];
        out[(size_t)(bbase + r) * ACT_DIM + a] = tanhf(raw);
    }
}

// ---------------------------------------------------------------------------
extern "C" void kernel_launch(void* const* d_in, const int* in_sizes, int n_in,
                              void* d_out, int out_size, void* d_ws, size_t ws_size,
                              hipStream_t stream)
{
    const float* obs   = (const float*)d_in[0];
    const float* mean  = (const float*)d_in[1];
    const float* stdv  = (const float*)d_in[2];
    const float* w1    = (const float*)d_in[3];
    const float* b1    = (const float*)d_in[4];
    const float* w2    = (const float*)d_in[5];
    const float* b2    = (const float*)d_in[6];
    const float* w3    = (const float*)d_in[7];
    const float* b3    = (const float*)d_in[8];
    const float* dec_w = (const float*)d_in[9];
    const float* dec_b = (const float*)d_in[10];

    int B = in_sizes[0] / OBS_DIM;   // 2048

    // workspace carve-up
    char* ws = (char*)d_ws;
    size_t off = 0;
    unsigned int* spk = (unsigned int*)(ws + off); off += (size_t)B * N_IN * 4;               // 20.97 MB
    _Float16* w1p = (_Float16*)(ws + off); off += (size_t)(N_IN / 32) * (HID / 16) * 512 * 2; // 1.31 MB
    _Float16* w2p = (_Float16*)(ws + off); off += W2P_BYTES;                                  // 128 KB
    _Float16* w3p = (_Float16*)(ws + off); off += W3P_BYTES;                                  // 40 KB
    (void)ws_size; (void)n_in; (void)out_size;

    {
        int total = B * N_IN;
        enc_pack_kernel<<<(total + 255) / 256, 256, 0, stream>>>(obs, mean, stdv, spk, total);
    }
    {
        int total = (N_IN / 32) * (HID / 16) * 512;
        pack_w_kernel<<<(total + 255) / 256, 256, 0, stream>>>(w1, w1p, N_IN, HID, total);
        total = (HID / 32) * (HID / 16) * 512;
        pack_w_kernel<<<(total + 255) / 256, 256, 0, stream>>>(w2, w2p, HID, HID, total);
        total = (HID / 32) * (OUTPOP / 16) * 512;
        pack_w_kernel<<<(total + 255) / 256, 256, 0, stream>>>(w3, w3p, HID, OUTPOP, total);
    }

    // dynamic LDS: 80K (spikes) + 8K + 8K + 5K + 128K (W2) + 40K (W3)
    //            = 275,456 B  (< 320 KB per WGP)
    size_t shmem = (size_t)BT * N_IN * 2 + 2 * (size_t)BT * HID * 2
                 + (size_t)BT * OUTPOP * 4 + W2P_BYTES + W3P_BYTES;
    snn_mlp_kernel<<<B / BT, 512, shmem, stream>>>(spk, w1p, w2p, w3p,
                                                   b1, b2, b3, dec_w, dec_b,
                                                   (float*)d_out);
}
